// GNN_13288628814645
// MI455X (gfx1250) — compile-verified, matrix operand order checked by reading, not run on hardware
//
#include <hip/hip_runtime.h>

typedef __attribute__((ext_vector_type(16))) _Float16 v16h;
typedef __attribute__((ext_vector_type(8)))  _Float16 v8h;
typedef __attribute__((ext_vector_type(8)))  float    v8f;

// Load one 16x32 f16 A-fragment for this lane: two contiguous 8-half (16B) runs
// at k-offsets +0 and +16 relative to p (p already includes hi*8).
__device__ __forceinline__ v16h ldfrag(const _Float16* __restrict__ p)
{
  v8h lo = *(const v8h*)p;
  v8h hp = *(const v8h*)(p + 16);
  v16h r;
#pragma unroll
  for (int i = 0; i < 8; ++i) { r[i] = lo[i]; r[i + 8] = hp[i]; }
  return r;
}

// ---------------------------------------------------------------------------
// WMMA GEMM: C[M,N] = A[M,Kp] @ W[K,N] (+bias)(+relu).
// A: f16, row-major, lda = Kp (K zero-padded to multiple of 32).
// Bp: weights pre-packed in B-fragment order:
//     Bp[((kb*(N/16)+nt)*32 + lane)*16 + i],  n = nt*16+(lane&15),
//     k = kb*32 + (lane>>4)*8 + i + (i&8 ? 8 : 0)
// One wave computes 64 rows x 16 cols (4 wmma tiles); block = 4 waves = 256 rows.
// Grid: (N/16, M/256). M must be multiple of 256, N of 16.
// ---------------------------------------------------------------------------
template<bool OF16, bool RELU>
__global__ void __launch_bounds__(128) k_gemm(
    const _Float16* __restrict__ A, const _Float16* __restrict__ Bp,
    const float* __restrict__ bias, void* __restrict__ Cout,
    int M, int N, int Kp)
{
  (void)M;
  int lane = threadIdx.x & 31;
  int w    = threadIdx.x >> 5;
  int nt   = blockIdx.x;
  int mq   = blockIdx.y * 4 + w;          // 64-row quad index
  int m0   = mq * 64;
  int ml   = lane & 15;
  int hi   = lane >> 4;
  int n    = nt * 16 + ml;
  int NT   = N >> 4;

  const _Float16* a0 = A + (size_t)(m0 + ml) * Kp + hi * 8;
  size_t astep = (size_t)16 * Kp;
  const _Float16* bp = Bp + ((size_t)nt * 32 + lane) * 16;
  size_t bstep = (size_t)NT * 512;

  v8f c0 = {}, c1 = {}, c2 = {}, c3 = {};
  for (int k0 = 0; k0 < Kp; k0 += 32) {
    v16h b = *(const v16h*)bp;
    bp += bstep;
    const _Float16* pa = a0 + k0;
    v16h x0 = ldfrag(pa);
    v16h x1 = ldfrag(pa + astep);
    v16h x2 = ldfrag(pa + 2 * astep);
    v16h x3 = ldfrag(pa + 3 * astep);
    c0 = __builtin_amdgcn_wmma_f32_16x16x32_f16(false, x0, false, b, (short)0, c0, false, false);
    c1 = __builtin_amdgcn_wmma_f32_16x16x32_f16(false, x1, false, b, (short)0, c1, false, false);
    c2 = __builtin_amdgcn_wmma_f32_16x16x32_f16(false, x2, false, b, (short)0, c2, false, false);
    c3 = __builtin_amdgcn_wmma_f32_16x16x32_f16(false, x3, false, b, (short)0, c3, false, false);
  }
  float bb = bias ? bias[n] : 0.f;
  float* Cf = (float*)Cout;
  _Float16* Ch = (_Float16*)Cout;
#pragma unroll
  for (int tt = 0; tt < 4; ++tt) {
    v8f c = tt == 0 ? c0 : (tt == 1 ? c1 : (tt == 2 ? c2 : c3));
#pragma unroll
    for (int r = 0; r < 8; ++r) {
      float v = c[r] + bb;
      if (RELU) v = fmaxf(v, 0.f);
      size_t row = (size_t)(m0 + tt * 16 + r + hi * 8);
      if (OF16) Ch[row * N + n] = (_Float16)v;
      else      Cf[row * N + n] = v;
    }
  }
}

// Pack f32 weight [K x N] into WMMA B-fragment order, zero-padding K to Kp.
__global__ void k_pack_w(const float* __restrict__ W, _Float16* __restrict__ out,
                         int K, int N, int Kp)
{
  int idx = blockIdx.x * blockDim.x + threadIdx.x;
  if (idx >= Kp * N) return;
  int i    = idx & 15;
  int lane = (idx >> 4) & 31;
  int rest = idx >> 9;                 // kb*NT + nt
  int NT   = N >> 4;
  int nt   = rest % NT;
  int kb   = rest / NT;
  int nn   = nt * 16 + (lane & 15);
  int hi   = lane >> 4;
  int k    = kb * 32 + hi * 8 + i + ((i & 8) ? 8 : 0);
  out[idx] = (k < K) ? (_Float16)W[(size_t)k * N + nn] : (_Float16)0.f;
}

// f32 [M x K] -> f16 [M x Kp], zero-padded along K.
__global__ void k_cvt_pad(const float* __restrict__ src, _Float16* __restrict__ dst,
                          int M, int K, int Kp)
{
  int i = blockIdx.x * blockDim.x + threadIdx.x;
  if (i >= M * Kp) return;
  int r = i / Kp, c = i - r * Kp;
  dst[i] = (c < K) ? (_Float16)src[(size_t)r * K + c] : (_Float16)0.f;
}

// ---------------------------------------------------------------------------
// Per-graph attention (one workgroup per graph; 256 edges, n_per nodes).
// ---------------------------------------------------------------------------
__global__ void k_attn(const _Float16* __restrict__ Q, const _Float16* __restrict__ Kf,
                       const _Float16* __restrict__ Vf, const _Float16* __restrict__ EP,
                       const int* __restrict__ src, const int* __restrict__ dst,
                       const int* __restrict__ msk, float* __restrict__ OUT, int n_per)
{
  extern __shared__ char smem_raw[];
  float* outacc = (float*)smem_raw;            // n_per*128
  float* alpha  = outacc + n_per * 128;        // 1024
  float* wgt    = alpha + 1024;                // 1024
  float* mmax   = wgt + 1024;                  // n_per*4
  float* dden   = mmax + n_per * 4;            // n_per*4
  int*   eS     = (int*)(dden + n_per * 4);
  int*   eD     = eS + 256;
  int*   eM     = eD + 256;

  int g = blockIdx.x, t = threadIdx.x;
  int nbase = g * n_per;
  int ebase = g * 256;

  { // P1: per-edge alpha
    int e = ebase + t;
    int s = src[e], d = dst[e], mk = msk[e];
    eS[t] = s; eD[t] = d - nbase; eM[t] = mk;
    const _Float16* qp = Q  + (size_t)d * 256;
    const _Float16* kp = Kf + (size_t)s * 256;
    const _Float16* ep = EP + (size_t)e * 256;
    for (int h = 0; h < 4; ++h) {
      float acc = 0.f;
      for (int c = 0; c < 64; ++c) {
        int cc = h * 64 + c;
        acc += (float)qp[cc] * ((float)kp[cc] + (float)ep[cc]);
      }
      alpha[t * 4 + h] = acc * 0.125f;
    }
  }
  __syncthreads();
  if (t < n_per * 4) { // P2: segment max + denom
    int nl = t >> 2, h = t & 3;
    float m = -1e30f;
    for (int e = 0; e < 256; ++e)
      if (eM[e] && eD[e] == nl) { float v = alpha[e * 4 + h]; m = v > m ? v : m; }
    float mm = (m > -1e29f) ? m : 0.f;
    float den = 0.f;
    for (int e = 0; e < 256; ++e)
      if (eM[e] && eD[e] == nl) den += __expf(alpha[e * 4 + h] - mm);
    mmax[t] = mm;
    dden[t] = den > 1e-16f ? den : 1e-16f;
  }
  __syncthreads();
  { // P3: softmax weights
    int nl = eD[t];
    for (int h = 0; h < 4; ++h) {
      float w = 0.f;
      if (eM[t]) w = __expf(alpha[t * 4 + h] - mmax[nl * 4 + h]) / dden[nl * 4 + h];
      wgt[t * 4 + h] = w;
    }
  }
  __syncthreads();
  for (int half = 0; half < 2; ++half) {   // P4/P5: aggregation
    for (int i = t; i < n_per * 128; i += 256) outacc[i] = 0.f;
    __syncthreads();
    for (int e = 0; e < 256; ++e) {
      if (!eM[e]) continue;                // uniform
      if (t < 128) {
        int ch = half * 128 + t;
        int h  = ch >> 6;
        int s  = eS[e];
        int nl = eD[e];
        float val = (float)Vf[(size_t)s * 256 + ch]
                  + (float)EP[(size_t)(ebase + e) * 256 + ch];
        outacc[nl * 128 + t] += wgt[e * 4 + h] * val;
      }
    }
    __syncthreads();
    for (int i = t; i < n_per * 128; i += 256) {
      int nl = i >> 7, ch = (i & 127) + half * 128;
      OUT[(size_t)(nbase + nl) * 256 + ch] = outacc[i];
    }
    __syncthreads();
  }
}

// Beta gate -> writes f16 HMID directly (only consumer is the WMMA transf GEMM)
__global__ void k_beta(const float* __restrict__ OUT, const float* __restrict__ XR,
                       const float* __restrict__ Wb, _Float16* __restrict__ HM16, int Nn)
{
  int node = blockIdx.x * 8 + (threadIdx.x >> 5);
  int lane = threadIdx.x & 31;
  if (node >= Nn) return;
  const float* o = OUT + (size_t)node * 256;
  const float* r = XR  + (size_t)node * 256;
  float s = 0.f;
  for (int c = lane; c < 256; c += 32) {
    float ov = o[c], rv = r[c];
    s += Wb[c] * ov + Wb[256 + c] * rv + Wb[512 + c] * (ov - rv);
  }
  for (int off = 16; off; off >>= 1) s += __shfl_xor(s, off, 32);
  float beta = 1.f / (1.f + __expf(-s));
  for (int c = lane; c < 256; c += 32)
    HM16[(size_t)node * 256 + c] = (_Float16)(beta * r[c] + (1.f - beta) * o[c]);
}

__global__ void k_zerof(float* __restrict__ p, int n)
{
  int i = blockIdx.x * blockDim.x + threadIdx.x;
  if (i < n) p[i] = 0.f;
}

__global__ void k_bn_stats(const float* __restrict__ T, float* __restrict__ stats, int Nn)
{
  __shared__ float ssum[64], ssq[64];
  int t = threadIdx.x;                 // 256
  if (t < 64) { ssum[t] = 0.f; ssq[t] = 0.f; }
  __syncthreads();
  int row0 = blockIdx.x * 64;
  int rend = row0 + 64; if (rend > Nn) rend = Nn;
  int ch = t & 63, r = t >> 6;
  float ls = 0.f, lq = 0.f;
  for (int rr = row0 + r; rr < rend; rr += 4) {
    float v = T[(size_t)rr * 64 + ch];
    ls += v; lq += v * v;
  }
  atomicAdd(&ssum[ch], ls);
  atomicAdd(&ssq[ch], lq);
  __syncthreads();
  if (t < 64) { atomicAdd(&stats[t], ssum[t]); atomicAdd(&stats[64 + t], ssq[t]); }
}

// BN apply: write f32 (topk/readout) and f16 (next layer GEMM A, Kp=64)
__global__ void k_bn_apply(const float* __restrict__ T, const float* __restrict__ stats,
                           const float* __restrict__ gg, const float* __restrict__ bb,
                           float* __restrict__ H, _Float16* __restrict__ Hh, int Nn)
{
  int i = blockIdx.x * blockDim.x + threadIdx.x;
  if (i >= Nn * 64) return;
  int ch = i & 63;
  float inv_n = 1.f / (float)Nn;
  float mu  = stats[ch] * inv_n;
  float var = stats[64 + ch] * inv_n - mu * mu;
  float v = (T[i] - mu) * rsqrtf(var + 1e-5f) * gg[ch] + bb[ch];
  H[i]  = v;
  Hh[i] = (_Float16)v;
}

// Per-graph TopK pooling; writes pooled rows both f32 and f16.
__global__ void k_topk(const float* __restrict__ H, const float* __restrict__ w,
                       float* __restrict__ HN, _Float16* __restrict__ HNh,
                       int* __restrict__ newid, int n_per, int kkeep)
{
  __shared__ float sc[64];
  __shared__ int   taken[64];
  __shared__ float bv[64];
  __shared__ int   bi[64];
  __shared__ int   selnode[32];
  __shared__ float selsc[32];
  __shared__ float wn;
  int g = blockIdx.x, t = threadIdx.x;   // 64 threads
  bv[t] = w[t] * w[t];
  __syncthreads();
  for (int s = 32; s; s >>= 1) { if (t < s) bv[t] += bv[t + s]; __syncthreads(); }
  if (t == 0) wn = sqrtf(bv[0]);
  __syncthreads();
  int nbase = g * n_per;
  if (t < n_per) {
    float d = 0.f;
    const float* hp = H + (size_t)(nbase + t) * 64;
    for (int c = 0; c < 64; ++c) d += hp[c] * w[c];
    sc[t] = tanhf(d / wn);
    taken[t] = 0;
    newid[nbase + t] = -1;
  }
  __syncthreads();
  for (int r = 0; r < kkeep; ++r) {
    bv[t] = (t < n_per && !taken[t]) ? sc[t] : -2.f;
    bi[t] = t;
    __syncthreads();
    for (int s = 32; s; s >>= 1) {
      if (t < s) {
        float v2 = bv[t + s]; int i2 = bi[t + s];
        if (v2 > bv[t] || (v2 == bv[t] && i2 < bi[t])) { bv[t] = v2; bi[t] = i2; }
      }
      __syncthreads();
    }
    if (t == 0) {
      int n = bi[0];
      taken[n] = 1; selnode[r] = n; selsc[r] = sc[n];
      newid[nbase + n] = g * kkeep + r;
    }
    __syncthreads();
  }
  for (int r = 0; r < kkeep; ++r) {
    float v = H[(size_t)(nbase + selnode[r]) * 64 + t] * selsc[r];
    HN [(size_t)(g * kkeep + r) * 64 + t] = v;
    HNh[(size_t)(g * kkeep + r) * 64 + t] = (_Float16)v;
  }
}

__global__ void k_remap(int* __restrict__ src, int* __restrict__ dst,
                        int* __restrict__ msk, const int* __restrict__ newid, int E)
{
  int e = blockIdx.x * blockDim.x + threadIdx.x;
  if (e >= E) return;
  int s = src[e], d = dst[e];
  int ns = newid[s], nd = newid[d];
  int ok = msk[e] && ns >= 0 && nd >= 0;
  src[e] = ok ? ns : 0;
  dst[e] = ok ? nd : 0;
  msk[e] = ok;
}

__global__ void k_init_edges(const int* __restrict__ ei, int* __restrict__ src,
                             int* __restrict__ dst, int* __restrict__ msk, int E)
{
  int e = blockIdx.x * blockDim.x + threadIdx.x;
  if (e < E) { src[e] = ei[e]; dst[e] = ei[E + e]; msk[e] = 1; }
}

__global__ void k_readout(const float* __restrict__ H, float* __restrict__ GB, int n_per)
{
  int g = blockIdx.x, c = threadIdx.x;   // 64 threads
  const float* hp = H + (size_t)g * n_per * 64;
  float mx = -1e30f, sm = 0.f;
  for (int n = 0; n < n_per; ++n) {
    float v = hp[n * 64 + c];
    mx = fmaxf(mx, v); sm += v;
  }
  GB[(size_t)g * 128 + c]      += mx;
  GB[(size_t)g * 128 + 64 + c] += sm / (float)n_per;
}

__global__ void k_final(const float* __restrict__ D2, const float* __restrict__ w,
                        const float* __restrict__ b, float* __restrict__ out)
{
  __shared__ float red[128];
  int r = blockIdx.x, t = threadIdx.x;   // 128 threads
  red[t] = D2[(size_t)r * 128 + t] * w[t];
  __syncthreads();
  for (int s = 64; s; s >>= 1) { if (t < s) red[t] += red[t + s]; __syncthreads(); }
  if (t == 0) out[r] = red[0] + b[0];
}

// ---------------------------------------------------------------------------
extern "C" void kernel_launch(void* const* d_in, const int* in_sizes, int n_in,
                              void* d_out, int out_size, void* d_ws, size_t ws_size,
                              hipStream_t stream)
{
  (void)in_sizes; (void)n_in; (void)out_size; (void)ws_size;
  const float* X  = (const float*)d_in[0];       // [32768,30]
  const float* EA = (const float*)d_in[1];       // [131072,11]
  const int*   EI = (const int*)d_in[2];         // [2,131072]
  auto P = [&](int i) { return (const float*)d_in[i]; };
  // insertion-order flatten: conv[5]{Wq,bq,Wk,bk,Wv,bv,We,Ws,bs,Wb} @4,
  // transf[5]{W,b} @54, bn[5]{g,b} @64, pool[2] @74, l1w..l3b @76..81

  const int N0 = 32768, E = 131072, NB = 512;

  char* ws = (char*)d_ws; size_t off = 0;
  auto carve = [&](size_t bytes) -> void* {
    void* p = ws + off; off += (bytes + 255) & ~(size_t)255; return p;
  };
  _Float16* Qh   = (_Float16*)carve((size_t)N0 * 256 * 2);
  _Float16* Kh   = (_Float16*)carve((size_t)N0 * 256 * 2);
  _Float16* Vh   = (_Float16*)carve((size_t)N0 * 256 * 2);
  _Float16* EPh  = (_Float16*)carve((size_t)E  * 256 * 2);
  _Float16* Xh   = (_Float16*)carve((size_t)N0 * 32 * 2);
  _Float16* EAh  = (_Float16*)carve((size_t)E  * 32 * 2);
  _Float16* HMIDh= (_Float16*)carve((size_t)N0 * 256 * 2);
  _Float16* HhA  = (_Float16*)carve((size_t)N0 * 64 * 2);
  _Float16* HhB  = (_Float16*)carve((size_t)N0 * 64 * 2);
  _Float16* GBh  = (_Float16*)carve((size_t)512 * 128 * 2);
  _Float16* D1h  = (_Float16*)carve((size_t)512 * 256 * 2);
  _Float16* WP   = (_Float16*)carve((size_t)256 * 256 * 2);   // packed weights
  float* XR   = (float*)carve((size_t)N0 * 256 * 4);
  float* OUTA = (float*)carve((size_t)N0 * 256 * 4);
  float* TRF  = (float*)carve((size_t)N0 * 64 * 4);
  float* HA   = (float*)carve((size_t)N0 * 64 * 4);
  float* HB   = (float*)carve((size_t)N0 * 64 * 4);
  int* SRC   = (int*)carve((size_t)E * 4);
  int* DST   = (int*)carve((size_t)E * 4);
  int* MSK   = (int*)carve((size_t)E * 4);
  int* NEWID = (int*)carve((size_t)N0 * 4);
  float* GBUF = (float*)carve(512 * 128 * 4);
  float* D2   = (float*)carve(512 * 128 * 4);
  float* BNS  = (float*)carve(512);

  auto pack = [&](const float* W, int K, int N, int Kp) {
    int tot = Kp * N;
    k_pack_w<<<(tot + 255) / 256, 256, 0, stream>>>(W, WP, K, N, Kp);
  };
  auto gemm = [&](const _Float16* A, const float* bias, void* C,
                  int M, int N, int Kp, bool of16, bool relu) {
    dim3 grid(N / 16, M / 256), block(128);
    if (of16 && relu)       k_gemm<true,  true ><<<grid, block, 0, stream>>>(A, WP, bias, C, M, N, Kp);
    else if (of16)          k_gemm<true,  false><<<grid, block, 0, stream>>>(A, WP, bias, C, M, N, Kp);
    else if (relu)          k_gemm<false, true ><<<grid, block, 0, stream>>>(A, WP, bias, C, M, N, Kp);
    else                    k_gemm<false, false><<<grid, block, 0, stream>>>(A, WP, bias, C, M, N, Kp);
  };

  // one-time input conversions
  k_cvt_pad<<<(N0 * 32 + 255) / 256, 256, 0, stream>>>(X,  Xh,  N0, 30, 32);
  k_cvt_pad<<<(E * 32 + 255) / 256, 256, 0, stream>>>(EA, EAh, E, 11, 32);
  k_init_edges<<<E / 256, 256, 0, stream>>>(EI, SRC, DST, MSK, E);
  k_zerof<<<(512 * 128 + 255) / 256, 256, 0, stream>>>(GBUF, 512 * 128);

  auto convblock = [&](const _Float16* hin, int cin, int Kp, int L, int Nn, int n_per,
                       float* houtF, _Float16* houtH) {
    int base = 4 + L * 10;
    pack(P(base + 0), cin, 256, Kp);
    gemm(hin, P(base + 1), Qh, Nn, 256, Kp, true, false);
    pack(P(base + 2), cin, 256, Kp);
    gemm(hin, P(base + 3), Kh, Nn, 256, Kp, true, false);
    pack(P(base + 4), cin, 256, Kp);
    gemm(hin, P(base + 5), Vh, Nn, 256, Kp, true, false);
    pack(P(base + 7), cin, 256, Kp);
    gemm(hin, P(base + 8), XR, Nn, 256, Kp, false, false);
    pack(P(base + 6), 11, 256, 32);
    gemm(EAh, nullptr, EPh, E, 256, 32, true, false);
    size_t smem = ((size_t)n_per * 128 + 2048 + (size_t)n_per * 8) * 4 + 3 * 256 * 4;
    k_attn<<<NB, 256, smem, stream>>>(Qh, Kh, Vh, EPh, SRC, DST, MSK, OUTA, n_per);
    k_beta<<<(Nn + 7) / 8, 256, 0, stream>>>(OUTA, XR, P(base + 9), HMIDh, Nn);
    pack(P(54 + L * 2), 256, 64, 256);
    gemm(HMIDh, P(55 + L * 2), TRF, Nn, 64, 256, false, true);
    k_zerof<<<1, 128, 0, stream>>>(BNS, 128);
    k_bn_stats<<<Nn / 64, 256, 0, stream>>>(TRF, BNS, Nn);
    k_bn_apply<<<(Nn * 64 + 255) / 256, 256, 0, stream>>>(
        TRF, BNS, P(64 + L * 2), P(65 + L * 2), houtF, houtH, Nn);
  };

  int n_per = 64, Nn = N0;
  convblock(Xh, 30, 32, 0, Nn, n_per, HA, HhA);
  float* Hcur = HA; float* Halt = HB;
  _Float16* Hhcur = HhA; _Float16* Hhalt = HhB;
  for (int i = 0; i < 4; ++i) {
    convblock(Hhcur, 64, 64, i + 1, Nn, n_per, Hcur, Hhcur);
    if (i % 2 == 0 || i == 3) {
      int kk = n_per / 2;
      k_topk<<<NB, 64, 0, stream>>>(Hcur, P(74 + i / 2), Halt, Hhalt, NEWID, n_per, kk);
      k_remap<<<E / 256, 256, 0, stream>>>(SRC, DST, MSK, NEWID, E);
      n_per = kk; Nn = NB * kk;
      { float* tp = Hcur; Hcur = Halt; Halt = tp; }
      { _Float16* tp = Hhcur; Hhcur = Hhalt; Hhalt = tp; }
      k_readout<<<NB, 64, 0, stream>>>(Hcur, GBUF, n_per);
    }
  }

  // dense head
  k_cvt_pad<<<(512 * 128 + 255) / 256, 256, 0, stream>>>(GBUF, GBh, 512, 128, 128);
  pack(P(76), 128, 256, 128);
  gemm(GBh, P(77), D1h, 512, 256, 128, true, true);
  pack(P(78), 256, 128, 256);
  gemm(D1h, P(79), D2, 512, 128, 256, false, true);
  k_final<<<512, 128, 0, stream>>>(D2, P(80), P(81), (float*)d_out);
}
